// SOFM1D_70755291234510
// MI455X (gfx1250) — matrix operand order repaired and checked by reading, not compiled, not run on hardware
//
#include <hip/hip_runtime.h>

typedef unsigned short u16;
typedef unsigned int   u32;
typedef unsigned long long u64;

typedef __attribute__((ext_vector_type(16))) __bf16 v16bf;
typedef __attribute__((ext_vector_type(8)))  float  f32x8;
typedef __attribute__((ext_vector_type(4)))  unsigned int u32x4;
typedef __attribute__((ext_vector_type(8)))  int    i32x8;
typedef __attribute__((ext_vector_type(4)))  int    i32x4;

#define BDIM 8192
#define DDIM 512
#define KDIM 16384

#define BM 128
#define BN 128
#define BK 32
#define NKT (DDIM / BK)    // 16 K-steps
#define NCB (KDIM / BN)    // 128 column blocks per row

// workspace layout (bytes)
#define OFF_XB   ((size_t)0)                                   // x bf16   [B][D]   8 MB
#define OFF_WT   (OFF_XB + (size_t)BDIM * DDIM * 2)            // wT bf16  [K][D]  16 MB
#define OFF_XSQ  (OFF_WT + (size_t)KDIM * DDIM * 2)            // ||x||^2  [B]
#define OFF_WSQ  (OFF_XSQ + (size_t)BDIM * 4)                  // ||w||^2  [K]
#define OFF_PART (OFF_WSQ + (size_t)KDIM * 4)                  // partial mins 8 MB

#if defined(__gfx1250__) && __has_builtin(__builtin_amdgcn_tensor_load_to_lds) && \
    __has_builtin(__builtin_amdgcn_s_wait_tensorcnt)
#define USE_TDM 1
#else
#define USE_TDM 0
#endif

struct __align__(8) MinPair { float v; int i; };

__device__ __forceinline__ u16 f2bf(float f) {
    u32 u = __float_as_uint(f);
    u32 r = u + 0x7FFFu + ((u >> 16) & 1u);   // round-to-nearest-even
    return (u16)(r >> 16);
}
__device__ __forceinline__ u32 pack2bf(float a, float b) {
    return (u32)f2bf(a) | ((u32)f2bf(b) << 16);
}

#if USE_TDM
// Issue one TDM 2D tile load (elements = 2 bytes): tile_d0 x tile_d1 from a
// row-major tensor (row length stride_d0 elements) into LDS at lds_off.
// D# layout per CDNA5 ISA.
__device__ __forceinline__ void tdm_load_2d(const void* gptr, u32 lds_off,
                                            u32 tensor_d0, u32 tensor_d1,
                                            u32 stride_d0,
                                            u32 tile_d0, u32 tile_d1) {
    u64 ga = (u64)(uintptr_t)gptr;
    u32x4 g0;
    g0[0] = 1u;                                  // count=1, user descriptor
    g0[1] = lds_off;                             // LDS byte address
    g0[2] = (u32)ga;                             // global_addr[31:0]
    g0[3] = (u32)((ga >> 32) & 0x01FFFFFFu) | (2u << 30);  // addr[56:32] | type=2
    i32x8 g1;
    g1[0] = (int)(1u << 16);                     // data_size=1 -> 2-byte elems
    g1[1] = (int)(tensor_d0 << 16);              // tensor_dim0[15:0]
    g1[2] = (int)((tensor_d0 >> 16) | (tensor_d1 << 16));
    g1[3] = (int)((tensor_d1 >> 16) | (tile_d0 << 16));   // tile_dim0
    g1[4] = (int)(tile_d1 & 0xFFFFu);            // tile_dim1 (tile_dim2=0)
    g1[5] = (int)stride_d0;                      // tensor_dim0_stride[31:0]
    g1[6] = 0;
    g1[7] = 0;
    i32x4 z4 = {0, 0, 0, 0};
#if __clang_major__ >= 23
    i32x8 z8 = {0, 0, 0, 0, 0, 0, 0, 0};
    __builtin_amdgcn_tensor_load_to_lds(g0, g1, z4, z4, z8, 0);
#else
    __builtin_amdgcn_tensor_load_to_lds(g0, g1, z4, z4, 0);
#endif
}
#endif

// ---- kernel 1: x -> bf16, per-row squared norm -------------------------------
__global__ void sofm_prep_x(const float* __restrict__ x, u16* __restrict__ xb,
                            float* __restrict__ xsq) {
    const int row = blockIdx.x;
    const int tid = threadIdx.x;                 // 256 threads, 2 floats each
    float2 v = ((const float2*)(x + (size_t)row * DDIM))[tid];
    ((u32*)xb)[(size_t)row * (DDIM / 2) + tid] = pack2bf(v.x, v.y);
    float s = v.x * v.x + v.y * v.y;
    for (int off = 16; off; off >>= 1) s += __shfl_xor(s, off, 32);
    __shared__ float red[8];
    if ((tid & 31) == 0) red[tid >> 5] = s;
    __syncthreads();
    if (tid == 0) {
        float t = 0.f;
        for (int i = 0; i < 8; ++i) t += red[i];
        xsq[row] = t;
    }
}

// ---- kernel 2: w -> wT bf16, per-column squared norm -------------------------
__global__ void sofm_prep_w(const float* __restrict__ w, u16* __restrict__ wt,
                            float* __restrict__ wsq) {
    const int k = blockIdx.x * 256 + threadIdx.x;
    float s = 0.f;
    for (int d0 = 0; d0 < DDIM; d0 += 32) {
        u32 buf[16];
#pragma unroll
        for (int dd = 0; dd < 32; dd += 2) {
            float a = w[(size_t)(d0 + dd)     * KDIM + k];
            float b = w[(size_t)(d0 + dd + 1) * KDIM + k];
            s += a * a + b * b;
            buf[dd >> 1] = pack2bf(a, b);
        }
        uint4* dst = (uint4*)(wt + (size_t)k * DDIM + d0);
        const uint4* src = (const uint4*)buf;
#pragma unroll
        for (int q = 0; q < 4; ++q) dst[q] = src[q];
    }
    wsq[k] = s;
}

// ---- kernel 3: WMMA GEMM + distance epilogue + fused partial argmin ----------
union FragU { uint4 q[2]; v16bf v; };

__global__ __launch_bounds__(256) void sofm_gemm(
        const u16* __restrict__ xb, const u16* __restrict__ wt,
        const float* __restrict__ xsq, const float* __restrict__ wsq,
        float* __restrict__ out, MinPair* __restrict__ part) {
    __shared__ u16 As[2][BM * BK];     // double-buffered [m][kk], 64B rows
    __shared__ u16 Bs[2][BN * BK];     // double-buffered [n][kk], 64B rows
    __shared__ float sXsq[BM];
    __shared__ float sWsq[BN];
    __shared__ MinPair sMin[BM][4];

    const int tid  = threadIdx.x;
    const int lane = tid & 31;
    const int wave = tid >> 5;         // 8 waves
    const int wm   = wave >> 2;        // 0..1 -> 64 rows each
    const int wn   = wave & 3;         // 0..3 -> 32 cols each
    const int half = lane >> 4;        // lane group (K-split per ISA layout)
    const int l16  = lane & 15;

    const int m0 = blockIdx.y * BM;
    const int n0 = blockIdx.x * BN;

    if (tid < 128) sXsq[tid] = xsq[m0 + tid];
    else           sWsq[tid - 128] = wsq[n0 + (tid - 128)];

    f32x8 acc[4][2] = {};

#if USE_TDM
    // ---- TDM staging: one DMA per tile, TENSORcnt-tracked, wave 0 issues ----
    if (wave == 0) {
        tdm_load_2d(xb + (size_t)m0 * DDIM, (u32)(uintptr_t)&As[0][0],
                    DDIM, BDIM, DDIM, BK, BM);
        tdm_load_2d(wt + (size_t)n0 * DDIM, (u32)(uintptr_t)&Bs[0][0],
                    DDIM, KDIM, DDIM, BK, BN);
    }
#else
    // ---- manual staging fallback: global -> VGPR -> LDS -----------------------
    {
#pragma unroll
        for (int i = 0; i < 2; ++i) {
            int c = tid + i * 256;
            int row = c >> 2, cc = c & 3;
            uint4 a = *(const uint4*)(xb + (size_t)(m0 + row) * DDIM + cc * 8);
            *(uint4*)(&As[0][0] + row * BK + cc * 8) = a;
            uint4 b = *(const uint4*)(wt + (size_t)(n0 + row) * DDIM + cc * 8);
            *(uint4*)(&Bs[0][0] + row * BK + cc * 8) = b;
        }
    }
#endif

    for (int kt = 0; kt < NKT; ++kt) {
        const int buf = kt & 1;
        // stage next tile into the other buffer (overwrites tile kt-1, which
        // the end-of-iteration barrier of kt-1 has already released)
        if (kt + 1 < NKT) {
            const int d1 = (kt + 1) * BK;
            const int nbuf = buf ^ 1;
#if USE_TDM
            if (wave == 0) {
                tdm_load_2d(xb + (size_t)m0 * DDIM + d1, (u32)(uintptr_t)&As[nbuf][0],
                            DDIM, BDIM, DDIM, BK, BM);
                tdm_load_2d(wt + (size_t)n0 * DDIM + d1, (u32)(uintptr_t)&Bs[nbuf][0],
                            DDIM, KDIM, DDIM, BK, BN);
            }
#else
#pragma unroll
            for (int i = 0; i < 2; ++i) {
                int c = tid + i * 256;
                int row = c >> 2, cc = c & 3;
                uint4 a = *(const uint4*)(xb + (size_t)(m0 + row) * DDIM + d1 + cc * 8);
                *(uint4*)(&As[nbuf][0] + row * BK + cc * 8) = a;
                uint4 b = *(const uint4*)(wt + (size_t)(n0 + row) * DDIM + d1 + cc * 8);
                *(uint4*)(&Bs[nbuf][0] + row * BK + cc * 8) = b;
            }
#endif
        }
#if USE_TDM
        if (wave == 0) {  // tile kt complete; tile kt+1 pair may stay in flight
            if (kt + 1 < NKT) __builtin_amdgcn_s_wait_tensorcnt(2);
            else              __builtin_amdgcn_s_wait_tensorcnt(0);
        }
#endif
        __syncthreads();

        // A fragment: lane holds row (l16), K chunks [half*8,+8) and [16+half*8,+8)
        v16bf af[4], bfr[2];
#pragma unroll
        for (int mt = 0; mt < 4; ++mt) {
            const u16* base = &As[buf][0] + (wm * 64 + mt * 16 + l16) * BK;
            FragU u;
            u.q[0] = *(const uint4*)(base + half * 8);
            u.q[1] = *(const uint4*)(base + 16 + half * 8);
            af[mt] = u.v;
        }
        // B fragment: lane holds col (l16), K chunks [half*8,+8) and [16+half*8,+8)
#pragma unroll
        for (int nt = 0; nt < 2; ++nt) {
            const u16* base = &Bs[buf][0] + (wn * 32 + nt * 16 + l16) * BK;
            FragU u;
            u.q[0] = *(const uint4*)(base + half * 8);
            u.q[1] = *(const uint4*)(base + 16 + half * 8);
            bfr[nt] = u.v;
        }
#pragma unroll
        for (int mt = 0; mt < 4; ++mt)
#pragma unroll
            for (int nt = 0; nt < 2; ++nt)
                acc[mt][nt] = __builtin_amdgcn_wmma_f32_16x16x32_bf16(
                    false, af[mt], false, bfr[nt], (short)0, acc[mt][nt],
                    false, false);
        __syncthreads();   // release buf for overwrite at iteration kt+1
    }

    // epilogue: dists = xsq + wsq - 2*acc ; NT stream stores ; fused argmin
#pragma unroll
    for (int mt = 0; mt < 4; ++mt) {
        const int rbase = wm * 64 + mt * 16 + half * 8;
#pragma unroll
        for (int r = 0; r < 8; ++r) {
            const int lrow = rbase + r;
            const int grow = m0 + lrow;
            const float xs = sXsq[lrow];
            const int c0 = wn * 32 + l16;
            const int c1 = c0 + 16;
            float v0 = xs + sWsq[c0] - 2.0f * acc[mt][0][r];
            float v1 = xs + sWsq[c1] - 2.0f * acc[mt][1][r];
            __builtin_nontemporal_store(v0, out + (size_t)grow * KDIM + n0 + c0);
            __builtin_nontemporal_store(v1, out + (size_t)grow * KDIM + n0 + c1);
            float mv = v0; int mi = n0 + c0;
            if (v1 < mv || (v1 == mv && (n0 + c1) < mi)) { mv = v1; mi = n0 + c1; }
            for (int off = 8; off; off >>= 1) {     // stays inside the 16-lane half
                float ov = __shfl_xor(mv, off, 32);
                int   oi = __shfl_xor(mi, off, 32);
                if (ov < mv || (ov == mv && oi < mi)) { mv = ov; mi = oi; }
            }
            if (l16 == 0) { sMin[lrow][wn].v = mv; sMin[lrow][wn].i = mi; }
        }
    }
    __syncthreads();
    if (tid < BM) {
        MinPair best = sMin[tid][0];
        for (int w4 = 1; w4 < 4; ++w4) {
            MinPair c = sMin[tid][w4];
            if (c.v < best.v || (c.v == best.v && c.i < best.i)) best = c;
        }
        part[(size_t)(m0 + tid) * NCB + blockIdx.x] = best;
    }
}

// ---- kernel 4: reduce per-block partials to one argmin per row ---------------
__global__ void sofm_argmin_final(const MinPair* __restrict__ part,
                                  float* __restrict__ idx_out) {
    const int b = blockIdx.x * blockDim.x + threadIdx.x;   // 0..B-1
    const MinPair* p = part + (size_t)b * NCB;
    float bv = p[0].v; int bi = p[0].i;
    for (int i = 1; i < NCB; ++i) {
        MinPair c = p[i];
        if (c.v < bv || (c.v == bv && c.i < bi)) { bv = c.v; bi = c.i; }
    }
    idx_out[b] = (float)bi;
}

extern "C" void kernel_launch(void* const* d_in, const int* in_sizes, int n_in,
                              void* d_out, int out_size, void* d_ws, size_t ws_size,
                              hipStream_t stream) {
    const float* x = (const float*)d_in[0];
    const float* w = (const float*)d_in[1];
    float* out = (float*)d_out;

    char* ws = (char*)d_ws;
    u16*     xb   = (u16*)(ws + OFF_XB);
    u16*     wt   = (u16*)(ws + OFF_WT);
    float*   xsq  = (float*)(ws + OFF_XSQ);
    float*   wsq  = (float*)(ws + OFF_WSQ);
    MinPair* part = (MinPair*)(ws + OFF_PART);

    sofm_prep_x<<<BDIM, 256, 0, stream>>>(x, xb, xsq);
    sofm_prep_w<<<KDIM / 256, 256, 0, stream>>>(w, wt, wsq);

    dim3 grid(KDIM / BN, BDIM / BM);   // 128 x 64 blocks
    sofm_gemm<<<grid, 256, 0, stream>>>(xb, wt, xsq, wsq, out, part);

    sofm_argmin_final<<<BDIM / 256, 256, 0, stream>>>(part,
                                                      out + (size_t)BDIM * KDIM);
}
